// HeteroGNN_51719996178797
// MI455X (gfx1250) — compile-verified, matrix operand order checked by reading, not run on hardware
//
#include <hip/hip_runtime.h>
#include <math.h>

typedef __attribute__((ext_vector_type(2)))  float  v2f;
typedef __attribute__((ext_vector_type(8)))  float  v8f;
typedef __attribute__((ext_vector_type(8)))  __bf16 v8bf;
typedef __attribute__((ext_vector_type(16))) __bf16 v16bf;

static constexpr int H_  = 256;
static constexpr int NP_ = 8192;
static constexpr int NJ_ = 8192;
static constexpr int EA_ = 262144;

// ---------- small utility kernels ----------

__global__ void zero_f32_kernel(float* __restrict__ p, int n) {
    int i = blockIdx.x * blockDim.x + threadIdx.x;
    if (i < n) p[i] = 0.0f;
}

__global__ void degree_kernel(const int* __restrict__ ei,
                              float* __restrict__ deg_src,
                              float* __restrict__ deg_dst, int ne) {
    int e = blockIdx.x * blockDim.x + threadIdx.x;
    if (e < ne) {
        atomicAdd(&deg_src[ei[e]], 1.0f);
        atomicAdd(&deg_dst[ei[ne + e]], 1.0f);
    }
}

__global__ void rsqrt_kernel(float* __restrict__ d, int n) {
    int i = blockIdx.x * blockDim.x + threadIdx.x;
    if (i < n) {
        float v = d[i];
        d[i] = (v > 0.0f) ? rsqrtf(v) : 0.0f;
    }
}

__global__ void init_bias_kernel(float* __restrict__ jemb,
                                 const float* __restrict__ bias, int n) {
    int i = blockIdx.x * blockDim.x + threadIdx.x;
    if (i < n) jemb[i] = bias[i & (H_ - 1)];
}

// one block (256 threads = 256 channels) per edge
__global__ void scatter_kernel(const int* __restrict__ ei,
                               const float* __restrict__ dis_s,
                               const float* __restrict__ dis_d,
                               const float* __restrict__ h,
                               float* __restrict__ jemb, int ne) {
    int e  = blockIdx.x;
    int ch = threadIdx.x;
    int r = ei[e];
    int c = ei[ne + e];
    float norm = dis_s[r] * dis_d[c];
    atomicAdd(&jemb[(size_t)c * H_ + ch], norm * h[(size_t)r * H_ + ch]);
}

// f32 -> (hi, lo) bf16 split:  x = hi + lo, lo captures the next 8 mantissa bits
__global__ void split_bf16_kernel(const float* __restrict__ x,
                                  __bf16* __restrict__ hi,
                                  __bf16* __restrict__ lo, int n) {
    int i = blockIdx.x * blockDim.x + threadIdx.x;
    if (i < n) {
        float v = x[i];
        __bf16 h = (__bf16)v;
        float r = v - (float)h;
        hi[i] = h;
        lo[i] = (__bf16)r;
    }
}

// ---------- WMMA GEMM (exact f32): h = P[NP,H] @ W[H,H] ----------

__global__ void gemm_h_kernel(const float* __restrict__ P,
                              const float* __restrict__ W,
                              float* __restrict__ Hout) {
    int wave = (blockIdx.x * blockDim.x + threadIdx.x) >> 5;
    int lane = threadIdx.x & 31;
    int tm = wave >> 4;           // NP/16 = 512 row tiles, H/16 = 16 col tiles
    int tn = wave & 15;
    int row0 = tm * 16, col0 = tn * 16;
    int l15  = lane & 15;
    int koff = (lane >> 4) << 1;  // lanes 0-15: K pair {0,1}; lanes 16-31: {2,3}

    const float* pa = P + (size_t)(row0 + l15) * H_ + koff;
    v8f acc = {};
    for (int k = 0; k < H_; k += 4) {
        v2f a = *(const v2f*)(pa + k);              // A: 16x4 f32 tile
        v2f b;                                      // B: 4x16 tile, B[k][n]=W[k][col0+n]
        b.x = W[(size_t)(k + koff)     * H_ + col0 + l15];
        b.y = W[(size_t)(k + koff + 1) * H_ + col0 + l15];
        acc = __builtin_amdgcn_wmma_f32_16x16x4_f32(
            false, a, false, b, (short)0, acc, false, false);
    }
    int mbase = (lane >> 4) << 3;   // C/D: lanes 0-15 -> M=v, lanes 16-31 -> M=8+v
    #pragma unroll
    for (int v = 0; v < 8; ++v)
        Hout[(size_t)(row0 + mbase + v) * H_ + col0 + l15] = acc[v];
}

// ---------- bf16x3 WMMA GEMM + sigmoid: out = sigmoid(P @ J^T) ----------
// 16-bit A fragment (16x32): lane = M row; lanes 0-15 hold K {0..7,16..23},
// lanes 16-31 hold K {8..15,24..31} -> two contiguous 16B chunks per lane.

union Frag16 { v16bf v; v8bf h[2]; };

__device__ __forceinline__ v16bf load_frag_bf16(const __bf16* __restrict__ rowbase,
                                                int kb) {
    Frag16 f;
    f.h[0] = *(const v8bf*)(rowbase + kb);
    f.h[1] = *(const v8bf*)(rowbase + kb + 16);
    return f.v;
}

__device__ __forceinline__ float sigmoidf_(float x) {
    return 1.0f / (1.0f + __expf(-x));
}

__global__ void scores_kernel(const __bf16* __restrict__ Phi,
                              const __bf16* __restrict__ Plo,
                              const __bf16* __restrict__ Jhi,
                              const __bf16* __restrict__ Jlo,
                              float* __restrict__ out) {
    int wave = (blockIdx.x * blockDim.x + threadIdx.x) >> 5;
    int lane = threadIdx.x & 31;
    int tm = wave >> 8;           // NP/32 = 256
    int tn = wave & 255;          // NJ/32 = 256
    int row0 = tm * 32, col0 = tn * 32;
    int l15  = lane & 15;
    int khalf = (lane >> 4) << 3;   // 0 or 8

    size_t ra0 = (size_t)(row0 + l15)      * H_;
    size_t ra1 = (size_t)(row0 + 16 + l15) * H_;
    size_t rb0 = (size_t)(col0 + l15)      * H_;   // B[k][n] = J[col0+n][k]
    size_t rb1 = (size_t)(col0 + 16 + l15) * H_;

    v8f acc00 = {}, acc01 = {}, acc10 = {}, acc11 = {};

    #pragma unroll 2
    for (int k = 0; k < H_; k += 32) {
        int kb = k + khalf;
        v16bf a0h = load_frag_bf16(Phi + ra0, kb);
        v16bf a0l = load_frag_bf16(Plo + ra0, kb);
        v16bf a1h = load_frag_bf16(Phi + ra1, kb);
        v16bf a1l = load_frag_bf16(Plo + ra1, kb);
        v16bf b0h = load_frag_bf16(Jhi + rb0, kb);
        v16bf b0l = load_frag_bf16(Jlo + rb0, kb);
        v16bf b1h = load_frag_bf16(Jhi + rb1, kb);
        v16bf b1l = load_frag_bf16(Jlo + rb1, kb);

        // bf16x3: hi*hi + hi*lo + lo*hi, f32 accumulate
        acc00 = __builtin_amdgcn_wmma_f32_16x16x32_bf16(false, a0h, false, b0h, (short)0, acc00, false, false);
        acc01 = __builtin_amdgcn_wmma_f32_16x16x32_bf16(false, a0h, false, b1h, (short)0, acc01, false, false);
        acc10 = __builtin_amdgcn_wmma_f32_16x16x32_bf16(false, a1h, false, b0h, (short)0, acc10, false, false);
        acc11 = __builtin_amdgcn_wmma_f32_16x16x32_bf16(false, a1h, false, b1h, (short)0, acc11, false, false);

        acc00 = __builtin_amdgcn_wmma_f32_16x16x32_bf16(false, a0h, false, b0l, (short)0, acc00, false, false);
        acc01 = __builtin_amdgcn_wmma_f32_16x16x32_bf16(false, a0h, false, b1l, (short)0, acc01, false, false);
        acc10 = __builtin_amdgcn_wmma_f32_16x16x32_bf16(false, a1h, false, b0l, (short)0, acc10, false, false);
        acc11 = __builtin_amdgcn_wmma_f32_16x16x32_bf16(false, a1h, false, b1l, (short)0, acc11, false, false);

        acc00 = __builtin_amdgcn_wmma_f32_16x16x32_bf16(false, a0l, false, b0h, (short)0, acc00, false, false);
        acc01 = __builtin_amdgcn_wmma_f32_16x16x32_bf16(false, a0l, false, b1h, (short)0, acc01, false, false);
        acc10 = __builtin_amdgcn_wmma_f32_16x16x32_bf16(false, a1l, false, b0h, (short)0, acc10, false, false);
        acc11 = __builtin_amdgcn_wmma_f32_16x16x32_bf16(false, a1l, false, b1h, (short)0, acc11, false, false);
    }

    int mbase = (lane >> 4) << 3;
    #pragma unroll
    for (int v = 0; v < 8; ++v) {
        size_t r0 = (size_t)(row0 + mbase + v) * NJ_;
        size_t r1 = (size_t)(row0 + 16 + mbase + v) * NJ_;
        out[r0 + col0 + l15]      = sigmoidf_(acc00[v]);
        out[r0 + col0 + 16 + l15] = sigmoidf_(acc01[v]);
        out[r1 + col0 + l15]      = sigmoidf_(acc10[v]);
        out[r1 + col0 + 16 + l15] = sigmoidf_(acc11[v]);
    }
}

// ---------- launch ----------

extern "C" void kernel_launch(void* const* d_in, const int* in_sizes, int n_in,
                              void* d_out, int out_size, void* d_ws, size_t ws_size,
                              hipStream_t stream) {
    const float* x_person = (const float*)d_in[0];
    const int*   ei       = (const int*)d_in[3];   // [2, EA] flattened: rows then cols
    const float* W_aj     = (const float*)d_in[6];
    const float* b_aj     = (const float*)d_in[7];
    float* out = (float*)d_out;

    // workspace layout (byte offsets, all 32KB-aligned regions)
    char* base = (char*)d_ws;
    float*  h     = (float*)(base);                                   // NP*H f32   (8 MB)
    float*  jemb  = (float*)(base + 8388608);                         // NJ*H f32   (8 MB)
    float*  deg_s = (float*)(base + 16777216);                        // NP f32
    float*  deg_d = (float*)(base + 16809984);                        // NJ f32
    __bf16* Phi   = (__bf16*)(base + 16842752);                       // NP*H bf16  (4 MB)
    __bf16* Plo   = (__bf16*)(base + 21037056);
    __bf16* Jhi   = (__bf16*)(base + 25231360);
    __bf16* Jlo   = (__bf16*)(base + 29425664);                       // ends ~33.6 MB

    // 1. zero degree buffers (deg_s, deg_d contiguous)
    zero_f32_kernel<<<(NP_ + NJ_ + 255) / 256, 256, 0, stream>>>(deg_s, NP_ + NJ_);
    // 2. degree counts
    degree_kernel<<<(EA_ + 255) / 256, 256, 0, stream>>>(ei, deg_s, deg_d, EA_);
    // 3. deg -> deg^-1/2 (0 where deg==0)
    rsqrt_kernel<<<(NP_ + NJ_ + 255) / 256, 256, 0, stream>>>(deg_s, NP_ + NJ_);
    // 4. h = x_person @ W_aj  (exact f32 WMMA)
    gemm_h_kernel<<<1024, 256, 0, stream>>>(x_person, W_aj, h);
    // 5. job_emb = broadcast(b_aj)
    init_bias_kernel<<<(NJ_ * H_) / 256, 256, 0, stream>>>(jemb, b_aj, NJ_ * H_);
    // 6. job_emb[col] += norm_e * h[row]
    scatter_kernel<<<EA_, 256, 0, stream>>>(ei, deg_s, deg_d, h, jemb, EA_);
    // 7. split P and job_emb into (hi, lo) bf16 pairs
    split_bf16_kernel<<<(NP_ * H_) / 256, 256, 0, stream>>>(x_person, Phi, Plo, NP_ * H_);
    split_bf16_kernel<<<(NJ_ * H_) / 256, 256, 0, stream>>>(jemb, Jhi, Jlo, NJ_ * H_);
    // 8. scores = sigmoid(P @ J^T) via bf16x3 WMMA (65536 waves, 32x32 tile each)
    scores_kernel<<<8192, 256, 0, stream>>>(Phi, Plo, Jhi, Jlo, out);
}
